// IncludeModel_37744172597284
// MI455X (gfx1250) — matrix operand order repaired and checked by reading, not compiled, not run on hardware
//
#include <hip/hip_runtime.h>
#include <hip/hip_bf16.h>

// ---------------------------------------------------------------------------
// Bidirectional LSTM (T=512, B=64, IN=1280, H=128) + temporal max-pool + head.
// CDNA5 / gfx1250, wave32.
//  - GEMMs via v_wmma_f32_16x16x32_bf16 with hi/lo bf16 split (fp32 accuracy).
//  - proj_gemm: double-buffered GLOBAL_LOAD_ASYNC_TO_LDS_B128 pipeline
//    (ASYNCcnt), async DMA of chunk k+1 overlapped with WMMAs of chunk k.
//  - lstm_scan: Tensor Data Mover prefetch of next step's gate tile
//    (tensor_load_to_lds, TENSORcnt) overlapped with recurrent WMMAs;
//    h state kept as bf16 hi/lo in LDS so the serial loop does no re-splits.
// ---------------------------------------------------------------------------

typedef __attribute__((ext_vector_type(16))) __bf16    v16bf;
typedef __attribute__((ext_vector_type(8)))  float     v8f;
typedef __attribute__((ext_vector_type(4)))  unsigned  u32x4;
typedef __attribute__((ext_vector_type(8)))  unsigned  u32x8;

#define T_STEPS 512
#define BATCH   64
#define INPUT   1280
#define HID     128
#define G4      512                      // 4*H
#define NCLS    263
#define KCHUNKS (INPUT / 32)             // 40
#define XPSZ    (BATCH * T_STEPS * G4)   // floats per direction (16.7M)

__device__ __forceinline__ float sigmoidf_(float x) {
    return 1.0f / (1.0f + __expf(-x));
}

__device__ __forceinline__ void split_bf(float v, __bf16& h, __bf16& l) {
    h = (__bf16)v;                    // RNE to bf16
    l = (__bf16)(v - (float)h);       // residual
}

__device__ __forceinline__ unsigned lds_off(const void* p) {
    // generic pointer to __shared__: low 32 bits are the LDS byte offset
    return (unsigned)(unsigned long long)p;
}

// Issue a 2D TDM load: 16 rows x 512 floats, row stride T*G4 floats -> LDS.
// D# group0/group1 packed per CDNA5 ISA §8.3/8.4; 2-group form (<=2D tensor).
__device__ __forceinline__ void tdm_load_tile(unsigned ldsByteOff, const float* gptr) {
    const unsigned long long ga = (unsigned long long)gptr;
    u32x4 g0;
    g0[0] = 1u;                                                 // count=1, user mode
    g0[1] = ldsByteOff;                                         // lds_addr
    g0[2] = (unsigned)ga;                                       // global_addr[31:0]
    g0[3] = (unsigned)((ga >> 32) & 0x1FFFFFFull) | (2u << 30); // addr[56:32] | type=2
    u32x8 g1;
    g1[0] = 2u << 16;                     // data_size = 4 bytes
    g1[1] = ((unsigned)G4) << 16;         // tensor_dim0[15:0] = 512
    g1[2] = 16u << 16;                    // tensor_dim1[15:0] = 16
    g1[3] = ((unsigned)G4) << 16;         // tile_dim0 = 512
    g1[4] = 16u;                          // tile_dim1 = 16 (tile_dim2 = 0)
    g1[5] = (unsigned)(T_STEPS * G4);     // tensor_dim0_stride[31:0] = 262144
    g1[6] = 0u;                           // stride[47:32]=0, dim1_stride lo=0
    g1[7] = 0u;
    asm volatile("tensor_load_to_lds %0, %1" :: "s"(g0), "s"(g1) : "memory");
}

// ---------------------------------------------------------------------------
// Kernel 1: input projection  xp[dir][b*T+t][0:512] = x_row . Wih^T + (bih+bhh)
// x is [B][T][IN] row-major, so GEMM row m = b*T+t maps directly onto x rows.
// 256 thr (8 waves), 64x64 tile. K-chunks of 32 double-buffered in LDS via
// async-to-LDS: wait(cur) -> barrier -> issue(next) -> compute(cur).
// ---------------------------------------------------------------------------
__global__ __launch_bounds__(256)
void proj_gemm(const float* __restrict__ x,
               const float* __restrict__ WihF, const float* __restrict__ bihF,
               const float* __restrict__ bhhF,
               const float* __restrict__ WihB, const float* __restrict__ bihB,
               const float* __restrict__ bhhB,
               float* __restrict__ ws)
{
    const int dir = blockIdx.z;
    const float* __restrict__ Wih = dir ? WihB : WihF;
    const float* __restrict__ bih = dir ? bihB : bihF;
    const float* __restrict__ bhh = dir ? bhhB : bhhF;
    float* __restrict__ xp = ws + (size_t)dir * XPSZ;

    const int mBase = blockIdx.x * 64;
    const int nBase = blockIdx.y * 64;

    __shared__ __align__(16) float As[2][64][32];   // 16 KB
    __shared__ __align__(16) float Bs[2][64][32];   // 16 KB

    const int tid  = threadIdx.x;
    const int lane = tid & 31;
    const int w    = tid >> 5;
    const int hi16 = lane >> 4;
    const int l16  = lane & 15;
    const int mt   = w >> 1;          // 0..3  (16-row tile)
    const int nt0  = (w & 1) * 2;     // 0 or 2 (two 16-col tiles)

    const unsigned long long xBase = (unsigned long long)x;
    const unsigned long long wBase = (unsigned long long)Wih;

    // per-thread fixed (row, quad) assignment: 2 x 16B per array per chunk
    auto issue_chunk = [&](int kc, int buf) {
        const int kBase = kc * 32;
        #pragma unroll
        for (int r = 0; r < 2; ++r) {
            const int idx = tid * 2 + r;        // 0..511
            const int row = idx >> 3;
            const int q   = idx & 7;
            const unsigned ldsA = lds_off(&As[buf][row][q * 4]);
            const unsigned offA = (unsigned)(((mBase + row) * INPUT + kBase + q * 4) * 4);
            asm volatile("global_load_async_to_lds_b128 %0, %1, %2"
                         :: "v"(ldsA), "v"(offA), "s"(xBase) : "memory");
            const unsigned ldsB = lds_off(&Bs[buf][row][q * 4]);
            const unsigned offB = (unsigned)(((nBase + row) * INPUT + kBase + q * 4) * 4);
            asm volatile("global_load_async_to_lds_b128 %0, %1, %2"
                         :: "v"(ldsB), "v"(offB), "s"(wBase) : "memory");
        }
    };

    v8f acc0 = {};
    v8f acc1 = {};

    issue_chunk(0, 0);

    for (int kc = 0; kc < KCHUNKS; ++kc) {
        const int buf = kc & 1;
        asm volatile("s_wait_asynccnt 0" ::: "memory");  // chunk kc landed
        __syncthreads();                                  // everyone done with buf^1
        if (kc + 1 < KCHUNKS) issue_chunk(kc + 1, buf ^ 1);  // overlap with compute

        // A fragment (16x32), ISA per-lane K mapping, hi/lo bf16 split
        v16bf aH, aL;
        #pragma unroll
        for (int i = 0; i < 16; ++i) {
            const int k = (i < 8 ? i : i + 8) + 8 * hi16;
            __bf16 h, l;
            split_bf(As[buf][mt * 16 + l16][k], h, l);
            aH[i] = h; aL[i] = l;
        }
        // B fragments (32x16) for the wave's two N-tiles
        v16bf b0H, b0L, b1H, b1L;
        #pragma unroll
        for (int i = 0; i < 16; ++i) {
            const int k = (i < 8 ? i : i + 8) + 8 * hi16;
            __bf16 h, l;
            split_bf(Bs[buf][nt0 * 16 + l16][k], h, l);       b0H[i] = h; b0L[i] = l;
            split_bf(Bs[buf][(nt0 + 1) * 16 + l16][k], h, l); b1H[i] = h; b1L[i] = l;
        }
        acc0 = __builtin_amdgcn_wmma_f32_16x16x32_bf16(false, aH, false, b0H, (short)0, acc0, false, false);
        acc0 = __builtin_amdgcn_wmma_f32_16x16x32_bf16(false, aH, false, b0L, (short)0, acc0, false, false);
        acc0 = __builtin_amdgcn_wmma_f32_16x16x32_bf16(false, aL, false, b0H, (short)0, acc0, false, false);
        acc1 = __builtin_amdgcn_wmma_f32_16x16x32_bf16(false, aH, false, b1H, (short)0, acc1, false, false);
        acc1 = __builtin_amdgcn_wmma_f32_16x16x32_bf16(false, aH, false, b1L, (short)0, acc1, false, false);
        acc1 = __builtin_amdgcn_wmma_f32_16x16x32_bf16(false, aL, false, b1H, (short)0, acc1, false, false);
    }

    // epilogue: add combined bias, scatter per C/D lane layout
    const int n0 = nBase + nt0 * 16 + l16;
    const int n1 = nBase + (nt0 + 1) * 16 + l16;
    const float bb0 = bih[n0] + bhh[n0];
    const float bb1 = bih[n1] + bhh[n1];
    #pragma unroll
    for (int v = 0; v < 8; ++v) {
        const int m = mBase + mt * 16 + v + 8 * hi16;
        xp[(size_t)m * G4 + n0] = acc0[v] + bb0;
        xp[(size_t)m * G4 + n1] = acc1[v] + bb1;
    }
}

// ---------------------------------------------------------------------------
// Kernel 2: LSTM scan, one workgroup per (direction, 16-batch slice).
// Whh resident in LDS as bf16 hi/lo (256KB). Wave w owns gate columns
// [16w,16w+16) of ALL four gates (tiles w, w+8, w+16, w+24): c-state, gate
// combine and temporal max stay in registers. h is stored ping-pong as bf16
// hi/lo (split once at write time; A-fragments are pure ds loads). Next
// step's xp tile is prefetched by the TDM into a 32KB LDS stage.
// LDS: 256K (Whh) + 16K (h hi/lo x2) + 32K (stage) = 304KB.
// ---------------------------------------------------------------------------
__global__ __launch_bounds__(256)
void lstm_scan(const float* __restrict__ WhhF, const float* __restrict__ WhhB,
               float* __restrict__ ws)
{
    const int dir = blockIdx.y;
    const int b0  = blockIdx.x * 16;
    const float* __restrict__ Whh = dir ? WhhB : WhhF;
    const float* __restrict__ xp  = ws + (size_t)dir * XPSZ;
    float* __restrict__ pooled    = ws + 2 * (size_t)XPSZ;

    __shared__ __bf16 WHi[G4 * HID];                 // 128 KB
    __shared__ __bf16 WLo[G4 * HID];                 // 128 KB
    __shared__ __bf16 hHi[2][16 * HID];              // 8 KB ping-pong
    __shared__ __bf16 hLo[2][16 * HID];              // 8 KB ping-pong
    __shared__ __align__(16) float stage[16 * G4];   // 32 KB TDM destination

    const int tid  = threadIdx.x;
    const int lane = tid & 31;
    const int w    = tid >> 5;
    const int hi16 = lane >> 4;
    const int l16  = lane & 15;
    const int hcol = w * 16 + l16;    // this lane's hidden column

    const unsigned stageOff = lds_off(&stage[0]);

    // kick off TDM for the first timestep's gate tile, overlap with Whh load
    if (w == 0) {
        const int tt0 = dir ? (T_STEPS - 1) : 0;
        tdm_load_tile(stageOff, xp + (size_t)(b0 * T_STEPS + tt0) * G4);
    }

    // load + split Whh into LDS once (B layout: [n][k] row-major)
    for (int i = tid; i < G4 * HID; i += 256) {
        __bf16 h, l;
        split_bf(Whh[i], h, l);
        WHi[i] = h; WLo[i] = l;
    }
    for (int i = tid; i < 16 * HID; i += 256) {
        hHi[0][i] = (__bf16)0.0f;
        hLo[0][i] = (__bf16)0.0f;
    }

    float cReg[8], mReg[8];
    #pragma unroll
    for (int v = 0; v < 8; ++v) { cReg[v] = 0.0f; mReg[v] = -3.4e38f; }

    if (w == 0) __builtin_amdgcn_s_wait_tensorcnt(0);
    __syncthreads();

    for (int t = 0; t < T_STEPS; ++t) {
        const int rb = t & 1;          // h read buffer
        const int wb = rb ^ 1;         // h write buffer

        // accumulators start from the staged xp tile: acc[j] = gate j's tile
        v8f acc[4];
        #pragma unroll
        for (int j = 0; j < 4; ++j) {
            const int n = (w + 8 * j) * 16 + l16;
            #pragma unroll
            for (int v = 0; v < 8; ++v)
                acc[j][v] = stage[(v + 8 * hi16) * G4 + n];
        }
        __syncthreads();   // stage consumed -> TDM may overwrite it

        if (w == 0 && t + 1 < T_STEPS) {
            const int tn = dir ? (T_STEPS - 2 - t) : (t + 1);
            tdm_load_tile(stageOff, xp + (size_t)(b0 * T_STEPS + tn) * G4);
        }

        #pragma unroll
        for (int kk = 0; kk < 4; ++kk) {
            const int k0 = kk * 32;
            v16bf aH, aL;
            #pragma unroll
            for (int i = 0; i < 16; ++i) {
                const int k = k0 + (i < 8 ? i : i + 8) + 8 * hi16;
                aH[i] = hHi[rb][l16 * HID + k];
                aL[i] = hLo[rb][l16 * HID + k];
            }
            #pragma unroll
            for (int j = 0; j < 4; ++j) {
                const int n = (w + 8 * j) * 16 + l16;
                v16bf bH, bL;
                #pragma unroll
                for (int i = 0; i < 16; ++i) {
                    const int k = k0 + (i < 8 ? i : i + 8) + 8 * hi16;
                    bH[i] = WHi[n * HID + k];
                    bL[i] = WLo[n * HID + k];
                }
                acc[j] = __builtin_amdgcn_wmma_f32_16x16x32_bf16(false, aH, false, bH, (short)0, acc[j], false, false);
                acc[j] = __builtin_amdgcn_wmma_f32_16x16x32_bf16(false, aH, false, bL, (short)0, acc[j], false, false);
                acc[j] = __builtin_amdgcn_wmma_f32_16x16x32_bf16(false, aL, false, bH, (short)0, acc[j], false, false);
            }
        }

        // fused gate nonlinearity + cell update + temporal max, all in regs;
        // h split to bf16 hi/lo exactly once, at write time
        #pragma unroll
        for (int v = 0; v < 8; ++v) {
            const float iv = sigmoidf_(acc[0][v]);
            const float fv = sigmoidf_(acc[1][v]);
            const float gv = tanhf(acc[2][v]);
            const float ov = sigmoidf_(acc[3][v]);
            const float c  = fv * cReg[v] + iv * gv;
            const float h  = ov * tanhf(c);
            cReg[v] = c;
            mReg[v] = fmaxf(mReg[v], h);
            __bf16 hh, hl;
            split_bf(h, hh, hl);
            const int idx = (v + 8 * hi16) * HID + hcol;
            hHi[wb][idx] = hh;
            hLo[wb][idx] = hl;
        }

        if (w == 0) __builtin_amdgcn_s_wait_tensorcnt(0);
        __syncthreads();   // h published + next xp tile landed
    }

    // pooled[b][dir*128 + hcol] = running max
    #pragma unroll
    for (int v = 0; v < 8; ++v) {
        const int m = v + 8 * hi16;
        pooled[(size_t)(b0 + m) * (2 * HID) + dir * HID + hcol] = mReg[v];
    }
}

// ---------------------------------------------------------------------------
// Kernel 3: head  out[b][c] = pooled[b][:] . W1[c][:] + b1[c]   (64 x 263)
// ---------------------------------------------------------------------------
__global__ __launch_bounds__(256)
void head_gemv(const float* __restrict__ pooled, const float* __restrict__ W1,
               const float* __restrict__ b1, float* __restrict__ out)
{
    const int idx = blockIdx.x * 256 + threadIdx.x;
    if (idx >= BATCH * NCLS) return;
    const int b = idx / NCLS;
    const int c = idx % NCLS;
    const float* pr = pooled + (size_t)b * (2 * HID);
    const float* wr = W1 + (size_t)c * (2 * HID);
    float s = b1[c];
    #pragma unroll 4
    for (int k = 0; k < 2 * HID; ++k) s += pr[k] * wr[k];
    out[idx] = s;
}

// ---------------------------------------------------------------------------
extern "C" void kernel_launch(void* const* d_in, const int* in_sizes, int n_in,
                              void* d_out, int out_size, void* d_ws, size_t ws_size,
                              hipStream_t stream)
{
    (void)in_sizes; (void)n_in; (void)out_size; (void)ws_size;
    const float* x    = (const float*)d_in[0];
    const float* WihF = (const float*)d_in[1];
    const float* WhhF = (const float*)d_in[2];
    const float* bihF = (const float*)d_in[3];
    const float* bhhF = (const float*)d_in[4];
    const float* WihB = (const float*)d_in[5];
    const float* WhhB = (const float*)d_in[6];
    const float* bihB = (const float*)d_in[7];
    const float* bhhB = (const float*)d_in[8];
    const float* W1   = (const float*)d_in[9];
    const float* b1   = (const float*)d_in[10];

    float* ws  = (float*)d_ws;
    float* out = (float*)d_out;

    // 1) input projections for both directions (fills ws[0 .. 2*XPSZ))
    dim3 g1(BATCH * T_STEPS / 64, G4 / 64, 2);   // 512 x 8 x 2 blocks
    proj_gemm<<<g1, 256, 0, stream>>>(x, WihF, bihF, bhhF, WihB, bihB, bhhB, ws);

    // 2) recurrent scans with fused max-pool -> pooled at ws + 2*XPSZ
    dim3 g2(BATCH / 16, 2);                      // 4 slices x 2 directions
    lstm_scan<<<g2, 256, 0, stream>>>(WhhF, WhhB, ws);

    // 3) classifier head
    const float* pooled = ws + 2 * (size_t)XPSZ;
    const int tot = BATCH * NCLS;
    head_gemv<<<(tot + 255) / 256, 256, 0, stream>>>(pooled, W1, b1, out);
}
